// MultiHeadAttention_76501957476829
// MI455X (gfx1250) — matrix operand order repaired and checked by reading, not compiled, not run on hardware
//
#include <hip/hip_runtime.h>
#include <hip/hip_bf16.h>
#include <math.h>

// Problem constants (from reference)
#define BB 2
#define TT 2048
#define HH 16
#define DH 64
#define DD 1024       // d_in == d_out == 1024
#define MM (BB * TT)  // 4096 token rows

typedef _Float16 half_t;
typedef __attribute__((ext_vector_type(8)))  _Float16 v8h;
typedef __attribute__((ext_vector_type(16))) _Float16 v16h;
typedef __attribute__((ext_vector_type(8)))  float    v8f;

// ---------------------------------------------------------------------------
// WMMA fragment helpers (wave32, gfx1250)
//
// A fragment 16x32 f16 (MxK): lane L holds row M = L&15,
//   element i: K = hi*16 + i, hi = L>>4.
//   Row-major (K contiguous) storage -> two 16B loads.
// B fragment 32x16 f16 (KxN): lane L holds col N = L&15, element i: K = hi*16+i.
//   N-major (K contiguous) storage -> one 32B load per lane.
// C/D fragment 16x16 f32: lane L holds col N = L&15, VGPR r holds M = r + hi*8.
// ---------------------------------------------------------------------------
static __device__ __forceinline__ v16h load_a_frag(const half_t* row, int hi) {
  v8h lo = *reinterpret_cast<const v8h*>(row + hi * 16);
  v8h hh = *reinterpret_cast<const v8h*>(row + hi * 16 + 8);
  return __builtin_shufflevector(lo, hh, 0,1,2,3,4,5,6,7,8,9,10,11,12,13,14,15);
}

static __device__ __forceinline__ v16h load_b_frag(const half_t* col, int hi) {
  return *reinterpret_cast<const v16h*>(col + hi * 16);
}

static __device__ __forceinline__ v8f wmma16(v16h a, v16h b, v8f c) {
  return __builtin_amdgcn_wmma_f32_16x16x32_f16(false, a, false, b, (short)0, c,
                                                false, false);
}

// ---------------------------------------------------------------------------
// CDNA5 async global->LDS copy (per-lane 16B), tracked by ASYNCcnt.
// VDST VGPR holds the LDS byte offset (low 32 bits of a flat shared address).
// ---------------------------------------------------------------------------
static __device__ __forceinline__ void async_copy_b128(half_t* lds_dst,
                                                       const half_t* gsrc) {
  uint32_t lds_off = (uint32_t)(uintptr_t)lds_dst;        // LDS byte offset
  uint64_t gaddr = (uint64_t)(uintptr_t)gsrc;
  asm volatile("global_load_async_to_lds_b128 %0, %1, off"
               :: "v"(lds_off), "v"(gaddr) : "memory");
}

static __device__ __forceinline__ void wait_async() {
  asm volatile("s_wait_asynccnt 0x0" ::: "memory");
}

// ---------------------------------------------------------------------------
// Kernel 1: f32 -> f16 elementwise (activations)
// ---------------------------------------------------------------------------
__global__ __launch_bounds__(256) void mha_cvt_f16(const float* __restrict__ s,
                                                   half_t* __restrict__ d, int n) {
  int i = blockIdx.x * blockDim.x + threadIdx.x;
  if (i < n) d[i] = (half_t)s[i];
}

// ---------------------------------------------------------------------------
// Kernel 2: transpose + convert weights: [K][N] f32 -> [N][K] f16 (B-friendly)
// ---------------------------------------------------------------------------
__global__ __launch_bounds__(256) void mha_cvt_wt(const float* __restrict__ wq,
                                                  const float* __restrict__ wk,
                                                  const float* __restrict__ wv,
                                                  const float* __restrict__ wo,
                                                  half_t* __restrict__ tq,
                                                  half_t* __restrict__ tk,
                                                  half_t* __restrict__ tv,
                                                  half_t* __restrict__ to) {
  const float* src = (blockIdx.z == 0) ? wq : (blockIdx.z == 1) ? wk
                    : (blockIdx.z == 2) ? wv : wo;
  half_t* dst = (blockIdx.z == 0) ? tq : (blockIdx.z == 1) ? tk
               : (blockIdx.z == 2) ? tv : to;
  __shared__ half_t tile[32][33];
  int tx = threadIdx.x & 31;
  int ty = threadIdx.x >> 5;
  int k0 = blockIdx.x * 32;
  int n0 = blockIdx.y * 32;
#pragma unroll
  for (int j = 0; j < 32; j += 8)
    tile[ty + j][tx] = (half_t)src[(size_t)(k0 + ty + j) * DD + n0 + tx];
  __syncthreads();
#pragma unroll
  for (int j = 0; j < 32; j += 8)
    dst[(size_t)(n0 + ty + j) * DD + k0 + tx] = tile[tx][ty + j];
}

// ---------------------------------------------------------------------------
// Shared staging helpers (block = 256 threads)
// ---------------------------------------------------------------------------
// B tile for GEMMs: 64(N) x 32(K) f16, N-major. One async B128 per thread.
static __device__ __forceinline__ void stage_b_tile(half_t* lds, const half_t* Wt,
                                                    int nbase, int k0, int tid) {
  const int n = tid >> 2, c = tid & 3;
  async_copy_b128(lds + n * 32 + c * 8,
                  Wt + (size_t)(nbase + n) * DD + k0 + c * 8);
}

// K tile for attention: 32(keys) x 64(dh) f16. One async B128 per thread.
static __device__ __forceinline__ void stage_k_tile(half_t* lds, const half_t* Kb,
                                                    int jbase, int tid) {
  const int r = tid >> 3, c = tid & 7;
  async_copy_b128(lds + r * 64 + c * 8,
                  Kb + (size_t)(jbase + r) * DH + c * 8);
}

// V^T tile for attention: 64(dh) x 32(keys) f16. One async B128 per thread.
static __device__ __forceinline__ void stage_v_tile(half_t* lds, const half_t* Vb,
                                                    int jbase, int tid) {
  const int r = tid >> 2, c = tid & 3;
  async_copy_b128(lds + r * 32 + c * 8,
                  Vb + (size_t)r * TT + jbase + c * 8);
}

// ---------------------------------------------------------------------------
// Kernel 3: QKV projections.  out = xh[4096x1024] @ Wt(sel).
// Block (256 thr / 8 waves) computes a 128x64 tile; the 64x32 B tile is
// staged in LDS via async copies (shared by all 8 waves), double buffered.
// Q,K stored [b,h,T,dh]; V stored transposed [b,h,dh,T].
// ---------------------------------------------------------------------------
__global__ __launch_bounds__(256) void mha_gemm_qkv(const half_t* __restrict__ xh,
                                                    const half_t* __restrict__ wtq,
                                                    const half_t* __restrict__ wtk,
                                                    const half_t* __restrict__ wtv,
                                                    half_t* __restrict__ Qh,
                                                    half_t* __restrict__ Kh,
                                                    half_t* __restrict__ Vt) {
  const int mode = blockIdx.z;
  const half_t* Wt = (mode == 0) ? wtq : (mode == 1) ? wtk : wtv;
  const int tid = threadIdx.x;
  const int lane = tid & 31;
  const int lq = lane & 15, hi = lane >> 4;
  const int w = tid >> 5;                          // wave 0..7
  const int mtile = (blockIdx.x >> 4) * 8 + w;     // 16-row tile, 0..255
  const int nbase = (blockIdx.x & 15) * 64;

  __shared__ half_t ldsB[2][64 * 32];              // 2 x 4KB

  v8f acc[4] = {};
  const half_t* arow = xh + (size_t)(mtile * 16 + lq) * DD;

  stage_b_tile(ldsB[0], Wt, nbase, 0, tid);
  v16h a_cur = load_a_frag(arow, hi);
  wait_async();
  __syncthreads();

  for (int kt = 0; kt < DD / 32; ++kt) {
    if (kt + 1 < DD / 32)
      stage_b_tile(ldsB[(kt + 1) & 1], Wt, nbase, (kt + 1) * 32, tid);
    const half_t* bcur = ldsB[kt & 1];
    v16h a = a_cur;
    if (kt + 1 < DD / 32) a_cur = load_a_frag(arow + (kt + 1) * 32, hi);
#pragma unroll
    for (int nt = 0; nt < 4; ++nt) {
      v16h b = load_b_frag(bcur + (nt * 16 + lq) * 32, hi);
      acc[nt] = wmma16(a, b, acc[nt]);
    }
    wait_async();
    __syncthreads();
  }

#pragma unroll
  for (int nt = 0; nt < 4; ++nt) {
    const int n = nbase + nt * 16 + lq;
    const int h = n >> 6, d = n & 63;
    if (mode == 2) {
      const int m0 = mtile * 16 + hi * 8;
      const int bi = m0 >> 11, t0 = m0 & (TT - 1);
      v8h o;
#pragma unroll
      for (int r = 0; r < 8; ++r) o[r] = (half_t)acc[nt][r];
      *reinterpret_cast<v8h*>(Vt + ((size_t)(bi * HH + h) * DH + d) * TT + t0) = o;
    } else {
      half_t* dst = (mode == 0) ? Qh : Kh;
#pragma unroll
      for (int r = 0; r < 8; ++r) {
        const int m = mtile * 16 + hi * 8 + r;
        const int bi = m >> 11, t = m & (TT - 1);
        dst[((size_t)(bi * HH + h) * TT + t) * DH + d] = (half_t)acc[nt][r];
      }
    }
  }
}

// ---------------------------------------------------------------------------
// Kernel 4: causal flash attention.  Block = 8 waves sharing (b,h) with
// consecutive 16-query tiles; K/V^T key-block tiles staged in LDS via async
// copies, double buffered.  Computes S^T = K*Q^T (softmax rows live in
// lanes); ctx^T += V^T * P^T with a one-shuffle P^T fragment rebuild.
// All waves iterate to the block-max causal length; out-of-range key blocks
// are numerically inert (masked -> p = 0, rescale = 1).
// ---------------------------------------------------------------------------
__global__ __launch_bounds__(256) void mha_attn(const half_t* __restrict__ Qh,
                                                const half_t* __restrict__ Kh,
                                                const half_t* __restrict__ Vt,
                                                half_t* __restrict__ ctxh) {
  const int tid = threadIdx.x;
  const int lane = tid & 31;
  const int lq = lane & 15, hi = lane >> 4;
  const int w = tid >> 5;
  const int qtg = blockIdx.x & 15;         // query-tile group (8 tiles)
  const int h   = (blockIdx.x >> 4) & 15;
  const int bi  = blockIdx.x >> 8;
  const int qt  = qtg * 8 + w;

  const half_t* Qb = Qh + (size_t)(bi * HH + h) * TT * DH;
  const half_t* Kb = Kh + (size_t)(bi * HH + h) * TT * DH;
  const half_t* Vb = Vt + (size_t)(bi * HH + h) * DH * TT;

  __shared__ half_t ldsK[2][32 * 64];      // 2 x 4KB: keys x dh
  __shared__ half_t ldsV[2][64 * 32];      // 2 x 4KB: dh x keys

  const int qbase = qt * 16;
  const int qg = qbase + lq;               // this lane's query (col of S^T)

  // Q^T B-fragments, invariant over key blocks (dh = 64 -> two K-chunks)
  const v16h qb0 = load_b_frag(Qb + (size_t)qg * DH + 0, hi);
  const v16h qb1 = load_b_frag(Qb + (size_t)qg * DH + 32, hi);

  v8f O[4] = {};
  float mrun = -INFINITY, lrun = 0.f;

  const int nb = qtg * 4 + 4;              // block-uniform #32-key blocks

  stage_k_tile(ldsK[0], Kb, 0, tid);
  stage_v_tile(ldsV[0], Vb, 0, tid);
  wait_async();
  __syncthreads();

  for (int jb = 0; jb < nb; ++jb) {
    if (jb + 1 < nb) {
      stage_k_tile(ldsK[(jb + 1) & 1], Kb, (jb + 1) * 32, tid);
      stage_v_tile(ldsV[(jb + 1) & 1], Vb, (jb + 1) * 32, tid);
    }
    const half_t* kc = ldsK[jb & 1];
    const half_t* vc = ldsV[jb & 1];
    const int jbase = jb * 32;

    // S^T tiles: keys [jbase, jbase+16) and [jbase+16, jbase+32)
    v8f s1 = {};
    s1 = wmma16(load_a_frag(kc + lq * 64 + 0, hi), qb0, s1);
    s1 = wmma16(load_a_frag(kc + lq * 64 + 32, hi), qb1, s1);
    v8f s2 = {};
    s2 = wmma16(load_a_frag(kc + (16 + lq) * 64 + 0, hi), qb0, s2);
    s2 = wmma16(load_a_frag(kc + (16 + lq) * 64 + 32, hi), qb1, s2);

    // scale + causal mask; softmax rows are lane columns (query qg)
    float p1[8], p2[8];
    float mt = -INFINITY;
#pragma unroll
    for (int r = 0; r < 8; ++r) {
      const int key1 = jbase + hi * 8 + r;
      const int key2 = key1 + 16;
      p1[r] = (key1 <= qg) ? s1[r] * 0.125f : -INFINITY;   // 1/sqrt(64)
      p2[r] = (key2 <= qg) ? s2[r] * 0.125f : -INFINITY;
      mt = fmaxf(mt, fmaxf(p1[r], p2[r]));
    }
    mt = fmaxf(mt, __shfl_xor(mt, 16, 32));
    const float mnew = fmaxf(mrun, mt);
    const float fscale = __expf(mrun - mnew);
    float rs = 0.f;
#pragma unroll
    for (int r = 0; r < 8; ++r) {
      p1[r] = __expf(p1[r] - mnew);
      p2[r] = __expf(p2[r] - mnew);
      rs += p1[r] + p2[r];
    }
    rs += __shfl_xor(rs, 16, 32);
    lrun = lrun * fscale + rs;
    mrun = mnew;
#pragma unroll
    for (int d = 0; d < 4; ++d) O[d] *= fscale;

    // Rebuild P^T as a B-fragment (lane=query, K=hi*16+i over 32 keys):
    // half the subtile lives in the partner half-wave -> shfl_xor(16).
    v16h pb;
#pragma unroll
    for (int i = 0; i < 8; ++i) {
      const float o1 = __shfl_xor(p1[i], 16, 32);
      const float o2 = __shfl_xor(p2[i], 16, 32);
      pb[i]     = (half_t)(hi ? o2 : p1[i]);   // key hi*16 + i
      pb[i + 8] = (half_t)(hi ? p2[i] : o1);   // key hi*16 + 8 + i
    }

    // ctx^T += V^T(16x32) * P^T(32x16) for each of 4 dh tiles
#pragma unroll
    for (int d = 0; d < 4; ++d) {
      v16h va = load_a_frag(vc + (d * 16 + lq) * 32, hi);
      O[d] = wmma16(va, pb, O[d]);
    }

    wait_async();
    __syncthreads();
  }

  const float inv = 1.f / lrun;
  const size_t row = (size_t)(bi * TT + qbase + lq);
#pragma unroll
  for (int d = 0; d < 4; ++d) {
    v8h o;
#pragma unroll
    for (int r = 0; r < 8; ++r) o[r] = (half_t)(O[d][r] * inv);
    *reinterpret_cast<v8h*>(ctxh + row * DD + h * 64 + d * 16 + hi * 8) = o;
  }
}

// ---------------------------------------------------------------------------
// Kernel 5: output projection  out = ctx @ Wo + bo  (f32 result), same
// LDS-staged double-buffered scheme as the QKV GEMM.
// ---------------------------------------------------------------------------
__global__ __launch_bounds__(256) void mha_gemm_out(const half_t* __restrict__ ctxh,
                                                    const half_t* __restrict__ wto,
                                                    const float* __restrict__ bo,
                                                    float* __restrict__ out) {
  const int tid = threadIdx.x;
  const int lane = tid & 31;
  const int lq = lane & 15, hi = lane >> 4;
  const int w = tid >> 5;
  const int mtile = (blockIdx.x >> 4) * 8 + w;
  const int nbase = (blockIdx.x & 15) * 64;

  __shared__ half_t ldsB[2][64 * 32];

  v8f acc[4] = {};
  const half_t* arow = ctxh + (size_t)(mtile * 16 + lq) * DD;

  stage_b_tile(ldsB[0], wto, nbase, 0, tid);
  v16h a_cur = load_a_frag(arow, hi);
  wait_async();
  __syncthreads();

  for (int kt = 0; kt < DD / 32; ++kt) {
    if (kt + 1 < DD / 32)
      stage_b_tile(ldsB[(kt + 1) & 1], wto, nbase, (kt + 1) * 32, tid);
    const half_t* bcur = ldsB[kt & 1];
    v16h a = a_cur;
    if (kt + 1 < DD / 32) a_cur = load_a_frag(arow + (kt + 1) * 32, hi);
#pragma unroll
    for (int nt = 0; nt < 4; ++nt) {
      v16h b = load_b_frag(bcur + (nt * 16 + lq) * 32, hi);
      acc[nt] = wmma16(a, b, acc[nt]);
    }
    wait_async();
    __syncthreads();
  }

#pragma unroll
  for (int nt = 0; nt < 4; ++nt) {
    const int n = nbase + nt * 16 + lq;
    const float bias = bo[n];
#pragma unroll
    for (int r = 0; r < 8; ++r) {
      const int m = mtile * 16 + hi * 8 + r;
      out[(size_t)m * DD + n] = acc[nt][r] + bias;
    }
  }
}

// ---------------------------------------------------------------------------
// Host launcher
// ---------------------------------------------------------------------------
extern "C" void kernel_launch(void* const* d_in, const int* in_sizes, int n_in,
                              void* d_out, int out_size, void* d_ws, size_t ws_size,
                              hipStream_t stream) {
  const float* x  = (const float*)d_in[0];
  const float* Wq = (const float*)d_in[1];
  const float* Wk = (const float*)d_in[2];
  const float* Wv = (const float*)d_in[3];
  const float* Wo = (const float*)d_in[4];
  const float* bo = (const float*)d_in[5];
  float* out = (float*)d_out;

  char* wsp = (char*)d_ws;
  size_t off = 0;
  auto carve = [&](size_t bytes) { void* p = wsp + off; off += bytes; return p; };
  half_t* xh   = (half_t*)carve((size_t)MM * DD * 2);   //  8 MiB
  half_t* wtq  = (half_t*)carve((size_t)DD * DD * 2);   //  2 MiB
  half_t* wtk  = (half_t*)carve((size_t)DD * DD * 2);
  half_t* wtv  = (half_t*)carve((size_t)DD * DD * 2);
  half_t* wto  = (half_t*)carve((size_t)DD * DD * 2);
  half_t* Qh   = (half_t*)carve((size_t)MM * DD * 2);   //  8 MiB
  half_t* Kh   = (half_t*)carve((size_t)MM * DD * 2);   //  8 MiB
  half_t* Vt   = (half_t*)carve((size_t)MM * DD * 2);   //  8 MiB
  half_t* ctxh = (half_t*)carve((size_t)MM * DD * 2);   //  8 MiB (48 total)
  (void)ws_size; (void)in_sizes; (void)n_in; (void)out_size;

  const int nelem = MM * DD;  // 4,194,304
  mha_cvt_f16<<<nelem / 256, 256, 0, stream>>>(x, xh, nelem);
  mha_cvt_wt<<<dim3(DD / 32, DD / 32, 4), 256, 0, stream>>>(Wq, Wk, Wv, Wo,
                                                            wtq, wtk, wtv, wto);
  // 32 M-blocks x 16 N-blocks = 512 blocks; z selects Q/K/V
  mha_gemm_qkv<<<dim3(512, 1, 3), 256, 0, stream>>>(xh, wtq, wtk, wtv, Qh, Kh, Vt);
  // 2 * 16 * 16 query-tile-groups = 512 blocks (8 waves each)
  mha_attn<<<512, 256, 0, stream>>>(Qh, Kh, Vt, ctxh);
  mha_gemm_out<<<512, 256, 0, stream>>>(ctxh, wto, bo, out);
}
